// AttentionLayerWithPrev_42356967473352
// MI455X (gfx1250) — matrix operand order repaired and checked by reading, not compiled, not run on hardware
//
#include <hip/hip_runtime.h>

// ---------------------------------------------------------------------------
// MI455X (gfx1250) residual-attention layer.
// All matmuls run on v_wmma_f32_16x16x32_bf16 (wave32, 16x16x32, f32 accum).
// Memory-bound problem (~1.2 GB HBM traffic vs ~52 GFLOP): each big
// [B,H,L,S] tensor is written once and read once; everything else lives in L2.
// This revision: fp32->bf16 via hardware convert (v_cvt_*_bf16_f32) instead of
// manual integer rounding, cutting VALU work in the A-fragment convert path.
// ---------------------------------------------------------------------------

typedef __attribute__((ext_vector_type(16))) __bf16        v16bf;
typedef __attribute__((ext_vector_type(8)))  float          v8f;
typedef __attribute__((ext_vector_type(8)))  unsigned short u16x8;
typedef __attribute__((ext_vector_type(4)))  float          f32x4;

static __device__ __forceinline__ unsigned short f2bf(float f) {
  union { __bf16 h; unsigned short u; } cv;
  cv.h = (__bf16)f;                    // hardware RNE convert on gfx1250
  return cv.u;
}

// ---------------------------------------------------------------------------
// Generic grouped WMMA GEMM:
//   C[g][m][n] = scale * sum_k A[g][m][k] * B[g][n][k]  (+ bias[n]) (+ prev[g][m][n])
// A: f32 or bf16, row-major, k contiguous.  B: bf16, stored as B^T ([N][K]).
// Group g = gb*Hgrp + gh with independent (batch, head) strides.
// Block = 128 threads = 4 waves; wave computes a 16x64 strip of C.
// ---------------------------------------------------------------------------
template<bool A_IS_F32, bool HAS_BIAS, bool HAS_PREV, bool OUT_BF16>
__global__ __launch_bounds__(128)
void wmma_gemm_kernel(const void* __restrict__ Aptr,
                      const unsigned short* __restrict__ Bptr,
                      const float* __restrict__ bias,
                      const float* __restrict__ prev,
                      void* __restrict__ Cptr,
                      int K, long lda, long ldb, long ldc,
                      int Hgrp,
                      long aSB, long aSH, long bSB, long bSH,
                      long cSB, long cSH, long pS, float scale)
{
  const int lane = threadIdx.x & 31;
  const int wave = threadIdx.x >> 5;
  const int l15  = lane & 15;
  const int hi   = lane >> 4;

  const int g  = blockIdx.z;
  const int gb = g / Hgrp;
  const int gh = g - gb * Hgrp;

  const int n0 = blockIdx.x * 64;
  const int m0 = blockIdx.y * 64 + wave * 16;

  const long aOff  = (long)gb * aSB + (long)gh * aSH + (long)(m0 + l15) * lda;
  const long bBase = (long)gb * bSB + (long)gh * bSH;
  const long cOff  = (long)gb * cSB + (long)gh * cSH;

  v8f acc[4] = {};

  for (int kb = 0; kb < K; kb += 32) {
    // A fragment (ISA 16-bit A layout): lane<16 -> K {kb..kb+7, kb+16..23},
    // lane>=16 -> shifted by +8.  Two 16B chunks per lane.
    union { v16bf bf; u16x8 u8[2]; unsigned short s[16]; } a;
    if (A_IS_F32) {
      const float* arow = (const float*)Aptr + aOff + kb + hi * 8;
      f32x4 f0 = *(const f32x4*)(arow);
      f32x4 f1 = *(const f32x4*)(arow + 4);
      f32x4 f2 = *(const f32x4*)(arow + 16);
      f32x4 f3 = *(const f32x4*)(arow + 20);
#pragma unroll
      for (int i = 0; i < 4; ++i) {
        a.bf[i]      = (__bf16)f0[i];
        a.bf[4 + i]  = (__bf16)f1[i];
        a.bf[8 + i]  = (__bf16)f2[i];
        a.bf[12 + i] = (__bf16)f3[i];
      }
    } else {
      const unsigned short* arow = (const unsigned short*)Aptr + aOff + kb + hi * 8;
      a.u8[0] = *(const u16x8*)(arow);
      a.u8[1] = *(const u16x8*)(arow + 16);
    }
#pragma unroll
    for (int t = 0; t < 4; ++t) {
      // B fragment: column n = n0+t*16+(lane&15); K {kb+8*hi..+7, kb+16+8*hi..+7} (matches A).
      const unsigned short* brow =
          Bptr + bBase + (long)(n0 + t * 16 + l15) * ldb + kb + hi * 8;
      union { v16bf bf; u16x8 u8[2]; } b;
      b.u8[0] = *(const u16x8*)(brow);
      b.u8[1] = *(const u16x8*)(brow + 16);
      acc[t] = __builtin_amdgcn_wmma_f32_16x16x32_bf16(
          false, a.bf, false, b.bf, (short)0, acc[t], false, false);
    }
  }

  // C fragment layout: VGPR r -> M = m0 + 8*hi + r, N = n0 + t*16 + (lane&15)
#pragma unroll
  for (int t = 0; t < 4; ++t) {
    const int n = n0 + t * 16 + l15;
    const float bv = HAS_BIAS ? bias[n] : 0.0f;
#pragma unroll
    for (int r = 0; r < 8; ++r) {
      const int m = m0 + hi * 8 + r;
      float v = acc[t][r] * scale + bv;
      if (HAS_PREV) v += prev[(long)g * pS + (long)m * ldc + n];
      const long ci = cOff + (long)m * ldc + n;
      if (OUT_BF16) ((unsigned short*)Cptr)[ci] = f2bf(v);
      else          ((float*)Cptr)[ci] = v;
    }
  }
}

// ---------------------------------------------------------------------------
// W [K,N] f32 -> WT [N,K] bf16 (coalesced write side; 4 MB per matrix)
// ---------------------------------------------------------------------------
__global__ __launch_bounds__(256)
void weight_transpose_kernel(const float* __restrict__ W,
                             unsigned short* __restrict__ WT, int K, int N) {
  long t = (long)blockIdx.x * blockDim.x + threadIdx.x;
  if (t >= (long)K * N) return;
  int k = (int)(t % K);
  int n = (int)(t / K);
  WT[t] = f2bf(W[(long)k * N + n]);
}

// Vt[b][h][d][s] = Vp[(b*S+s)*(H*DV) + h*DV + d]   (bf16 -> bf16)
__global__ __launch_bounds__(256)
void v_transpose_kernel(const unsigned short* __restrict__ Vp,
                        unsigned short* __restrict__ Vt,
                        int S, int Hh, int Dv) {
  long t = (long)blockIdx.x * blockDim.x + threadIdx.x;
  int s = (int)(t % S); long r = t / S;
  int d = (int)(r % Dv); r /= Dv;
  int h = (int)(r % Hh);
  int b = (int)(r / Hh);
  Vt[t] = Vp[((long)b * S + s) * (long)(Hh * Dv) + h * Dv + d];
}

// ---------------------------------------------------------------------------
// Row softmax with mask: one wave32 per row of [B*H*L] x S.
// attn = softmax(mask ? -1e9 : scores) ; scores output remains unmasked.
// ---------------------------------------------------------------------------
__global__ __launch_bounds__(256)
void softmax_rows_kernel(const float* __restrict__ scores,
                         const unsigned char* __restrict__ mask,
                         float* __restrict__ attn, int Hh, int L, int S)
{
  const int lane = threadIdx.x & 31;
  const int wave = threadIdx.x >> 5;
  const long row = (long)blockIdx.x * 8 + wave;
  const int  l  = (int)(row % L);
  const long bh = row / L;
  const int  b  = (int)(bh / Hh);

  const float* src = scores + row * (long)S;
  const unsigned char* mrow = mask + ((long)b * L + l) * (long)S;
  float* dst = attn + row * (long)S;

  float v[32];
  float mx = -3.0e38f;
#pragma unroll
  for (int j = 0; j < 8; ++j) {
    const int s = j * 128 + lane * 4;
    f32x4 x = *(const f32x4*)(src + s);
    uchar4 mk = *(const uchar4*)(mrow + s);
    v[j*4+0] = mk.x ? -1.0e9f : x[0];
    v[j*4+1] = mk.y ? -1.0e9f : x[1];
    v[j*4+2] = mk.z ? -1.0e9f : x[2];
    v[j*4+3] = mk.w ? -1.0e9f : x[3];
    mx = fmaxf(mx, fmaxf(fmaxf(v[j*4+0], v[j*4+1]), fmaxf(v[j*4+2], v[j*4+3])));
  }
#pragma unroll
  for (int off = 16; off > 0; off >>= 1) mx = fmaxf(mx, __shfl_xor(mx, off, 32));
  float sum = 0.0f;
#pragma unroll
  for (int i = 0; i < 32; ++i) { v[i] = __expf(v[i] - mx); sum += v[i]; }
#pragma unroll
  for (int off = 16; off > 0; off >>= 1) sum += __shfl_xor(sum, off, 32);
  const float inv = 1.0f / sum;
#pragma unroll
  for (int j = 0; j < 8; ++j) {
    const int s = j * 128 + lane * 4;
    f32x4 o;
    o[0] = v[j*4+0] * inv; o[1] = v[j*4+1] * inv;
    o[2] = v[j*4+2] * inv; o[3] = v[j*4+3] * inv;
    *(f32x4*)(dst + s) = o;
  }
}

// ---------------------------------------------------------------------------
extern "C" void kernel_launch(void* const* d_in, const int* in_sizes, int n_in,
                              void* d_out, int out_size, void* d_ws, size_t ws_size,
                              hipStream_t stream) {
  (void)in_sizes; (void)n_in; (void)out_size; (void)ws_size;

  const float* queries = (const float*)d_in[0];
  const float* keys    = (const float*)d_in[1];
  const float* values  = (const float*)d_in[2];
  const float* prev    = (const float*)d_in[3];
  const unsigned char* mask = (const unsigned char*)d_in[4];   // bool -> 1 byte
  const float* Wq = (const float*)d_in[5];
  const float* bq = (const float*)d_in[6];
  const float* Wk = (const float*)d_in[7];
  const float* bk = (const float*)d_in[8];
  const float* Wv = (const float*)d_in[9];
  const float* bv = (const float*)d_in[10];
  const float* Wo = (const float*)d_in[11];
  const float* bo = (const float*)d_in[12];

  constexpr int  Bn = 4, L = 1024, S = 1024, D = 1024, Hh = 16, DK = 64, DV = 64;
  constexpr long MR = (long)Bn * L;                 // 4096 projection rows

  // Workspace carve (bf16 elements): 48 MB total.
  unsigned short* WqT = (unsigned short*)d_ws;      // 1M each
  unsigned short* WkT = WqT + (long)D * D;
  unsigned short* WvT = WkT + (long)D * D;
  unsigned short* WoT = WvT + (long)D * D;
  unsigned short* Qp  = WoT + (long)D * D;          // 4M each
  unsigned short* Kp  = Qp  + MR * D;
  unsigned short* Vp  = Kp  + MR * D;
  unsigned short* Vt  = Vp  + MR * D;
  unsigned short* Obf = Vt  + MR * D;

  // d_out = [out | attn | scores]
  float* outF    = (float*)d_out;
  float* attnF   = outF  + MR * D;
  float* scoresF = attnF + (long)Bn * Hh * L * S;

  // 1) weights -> bf16 transposed [N][K]
  {
    const long tot = (long)D * D;
    const int  blocks = (int)((tot + 255) / 256);
    weight_transpose_kernel<<<blocks, 256, 0, stream>>>(Wq, WqT, D, D);
    weight_transpose_kernel<<<blocks, 256, 0, stream>>>(Wk, WkT, D, D);
    weight_transpose_kernel<<<blocks, 256, 0, stream>>>(Wv, WvT, D, D);
    weight_transpose_kernel<<<blocks, 256, 0, stream>>>(Wo, WoT, D, D);
  }

  const dim3 blk(128);

  // 2) QKV projections: [4096,1024] x [1024,1024] + bias -> bf16
  {
    dim3 grid(D / 64, (unsigned)(MR / 64), 1);
    wmma_gemm_kernel<true, true, false, true><<<grid, blk, 0, stream>>>(
        queries, WqT, bq, nullptr, Qp, D, D, D, D,
        1, 0, 0, 0, 0, 0, 0, 0, 1.0f);
    wmma_gemm_kernel<true, true, false, true><<<grid, blk, 0, stream>>>(
        keys, WkT, bk, nullptr, Kp, D, D, D, D,
        1, 0, 0, 0, 0, 0, 0, 0, 1.0f);
    wmma_gemm_kernel<true, true, false, true><<<grid, blk, 0, stream>>>(
        values, WvT, bv, nullptr, Vp, D, D, D, D,
        1, 0, 0, 0, 0, 0, 0, 0, 1.0f);
  }

  // 3) V -> Vt[b][h][d][s] so attn@V B-fragments are contiguous
  {
    const long tot = (long)Bn * Hh * DV * S;
    v_transpose_kernel<<<(int)((tot + 255) / 256), 256, 0, stream>>>(Vp, Vt, S, Hh, DV);
  }

  // 4) scores[b,h] = Q_bh (LxDK) * K_bh^T * (1/8) + prev_logits  -> f32
  {
    dim3 grid(S / 64, L / 64, Bn * Hh);
    wmma_gemm_kernel<false, false, true, false><<<grid, blk, 0, stream>>>(
        Qp, Kp, nullptr, prev, scoresF, DK,
        (long)Hh * DK, (long)Hh * DK, (long)S,
        Hh,
        (long)L * D, (long)DK,            // A group strides (b, h)
        (long)S * D, (long)DK,            // B group strides
        (long)Hh * L * S, (long)L * S,    // C group strides
        (long)L * S, 0.125f);
  }

  // 5) masked softmax rows -> attn (f32)
  {
    const long rows = (long)Bn * Hh * L;
    softmax_rows_kernel<<<(int)(rows / 8), 256, 0, stream>>>(scoresF, mask, attnF, Hh, L, S);
  }

  // 6) O[b,l,h*64+d] = attn_bh (LxS, f32->bf16 on load) * Vt_bh (DVxS)
  {
    dim3 grid(DV / 64, L / 64, Bn * Hh);
    wmma_gemm_kernel<true, false, false, true><<<grid, blk, 0, stream>>>(
        attnF, Vt, nullptr, nullptr, Obf, S,
        (long)S, (long)S, (long)D,
        Hh,
        (long)Hh * L * S, (long)L * S,    // A group strides
        (long)Hh * DV * S, (long)DV * S,  // B group strides
        (long)L * D, (long)DV,            // C group strides
        0, 1.0f);
  }

  // 7) out = O [4096,1024] x Wo + bo -> f32
  {
    dim3 grid(D / 64, (unsigned)(MR / 64), 1);
    wmma_gemm_kernel<false, true, false, false><<<grid, blk, 0, stream>>>(
        Obf, WoT, bo, nullptr, outF, D, D, D, D,
        1, 0, 0, 0, 0, 0, 0, 0, 1.0f);
  }
}
